// MultiHeadAttention_8297876816038
// MI455X (gfx1250) — matrix operand order repaired and checked by reading, not compiled, not run on hardware
//
#include <hip/hip_runtime.h>

typedef _Float16 h16;
typedef __attribute__((ext_vector_type(16))) _Float16 v16h;
typedef __attribute__((ext_vector_type(8)))  _Float16 v8h;
typedef __attribute__((ext_vector_type(8)))  float    v8f;
typedef __attribute__((ext_vector_type(4)))  unsigned int v4u;
typedef __attribute__((ext_vector_type(8)))  int v8i;
typedef __attribute__((ext_vector_type(4)))  int v4i;

#define D_MODEL 1024
#define N_HEADS 16
#define D_K     64
#define B_SZ    4
#define SEQ     2048
#define M_ROWS  (B_SZ*SEQ)   // 8192
#define KSTEP   64           // GEMM k-tile staged in LDS

#if defined(__has_builtin)
#  if __has_builtin(__builtin_amdgcn_tensor_load_to_lds) && \
      __has_builtin(__builtin_amdgcn_s_wait_tensorcnt)
#    define HAVE_TDM 1
#  endif
#endif
#ifndef HAVE_TDM
#  define HAVE_TDM 0
#endif

// ---------------------------------------------------------------------------
// helpers
// ---------------------------------------------------------------------------
__device__ __forceinline__ v16h combine8(v8h lo, v8h hi) {
  v16h r;
#pragma unroll
  for (int i = 0; i < 8; ++i) { r[i] = lo[i]; r[i + 8] = hi[i]; }
  return r;
}

__device__ __forceinline__ v8f wmma16(v16h a, v16h b, v8f c) {
  // D = A(16x32 f16) * B(32x16 f16) + C(16x16 f32)
  return __builtin_amdgcn_wmma_f32_16x16x32_f16(
      /*neg_a=*/false, a, /*neg_b=*/false, b,
      /*c_mod=*/(short)0, c, /*reuse_a=*/false, /*reuse_b=*/false);
}

#if HAVE_TDM
// Issue a TDM 2-D tile load (f16 elements) into LDS.
// D# packing per CDNA5 ISA ch.8.3/8.4: group0 = {count|flags, lds_addr,
// global_addr lo, global_addr hi | type=2}; group1 = {data_size, tensor dims,
// tile dims, dim0 stride}; groups 2/3 zero (<=2-D tensor).
// This toolchain's builtin takes 6 args (g0,g1,g2,g3,extra_g,cpol).
__device__ __forceinline__ void tdm_load_2d(unsigned int lds_byte_off,
                                            unsigned long long gaddr,
                                            unsigned int tensor_d0,
                                            unsigned int tensor_d1,
                                            unsigned int tile_d0,
                                            unsigned int tile_d1,
                                            unsigned long long stride0) {
  v4u g0;
  g0[0] = 1u;                                        // count=1, user descriptor
  g0[1] = lds_byte_off;                              // lds_addr
  g0[2] = (unsigned int)gaddr;                       // global_addr[31:0]
  g0[3] = (unsigned int)((gaddr >> 32) & 0x01FFFFFFu) | (2u << 30);  // [56:32]|type
  v8i g1;
  g1[0] = (int)(1u << 16);                           // data_size = 2 bytes
  g1[1] = (int)((tensor_d0 & 0xFFFFu) << 16);        // tensor_dim0[15:0]
  g1[2] = (int)((tensor_d0 >> 16) | ((tensor_d1 & 0xFFFFu) << 16));
  g1[3] = (int)((tensor_d1 >> 16) | (tile_d0 << 16));  // dim1 hi | tile_dim0
  g1[4] = (int)tile_d1;                              // tile_dim1 (tile_dim2=0)
  g1[5] = (int)(unsigned int)(stride0 & 0xFFFFFFFFu);  // dim0 stride lo
  g1[6] = (int)(unsigned int)(stride0 >> 32);        // dim0 stride hi
  g1[7] = 0;
  v4i gz4 = {0, 0, 0, 0};
  v8i gz8 = {0, 0, 0, 0, 0, 0, 0, 0};
  __builtin_amdgcn_tensor_load_to_lds(g0, g1, gz4, gz4, gz8, 0);
}
#endif

// ---------------------------------------------------------------------------
// fp32 -> fp16 conversion (grid-stride)
// ---------------------------------------------------------------------------
__global__ void cvt_f32_to_f16(const float* __restrict__ in,
                               h16* __restrict__ out, int n) {
  int i = blockIdx.x * blockDim.x + threadIdx.x;
  int stride = gridDim.x * blockDim.x;
  for (; i < n; i += stride) out[i] = (h16)in[i];
}

// ---------------------------------------------------------------------------
// GEMM: Y[m,n] = sum_k X[m,k] * W[n,k]        (K = N = D_MODEL = 1024)
// Block = 8 waves; wave w owns rows m0..m0+15, block owns 64 columns.
// The 64x64 f16 B tile (shared by all 8 waves) is staged in LDS by the
// Tensor Data Mover (double-buffered); A fragments are direct global loads.
// All 10 fragment loads of a k-tile are hoisted ahead of the 8-WMMA chain so
// the scheduler can clause the ds/global loads and retire waits incrementally.
// mode 0: outH row-major [M, D_MODEL]
// mode 1: outH[((b*16+h)*SEQ + s)*64 + d]          (Q / K head-split)
// mode 2: outH[((b*16+h)*64 + d)*SEQ + s]          (V transposed for P*V)
// mode 3: outF row-major [M, D_MODEL] fp32 (final output)
// ---------------------------------------------------------------------------
__global__ __launch_bounds__(256) void gemm_xwT(
    const h16* __restrict__ X, const h16* __restrict__ W,
    h16* __restrict__ outH, float* __restrict__ outF, int mode) {
  __shared__ __align__(16) h16 Bsh[2][64][KSTEP];    // 16 KB, double buffered

  const int tid  = threadIdx.x;
  const int lane = tid & 31;
  const int wave = tid >> 5;
  const int l16  = lane & 15;
  const int hi   = (lane >> 4) & 1;   // lane-half (K split per WMMA layout)

  const int m0 = blockIdx.y * 128 + wave * 16;
  const int n0 = blockIdx.x * 64;

  const h16* arow = X + (size_t)(m0 + l16) * D_MODEL;

  v8f acc[4];
  const v8f zero = {0.f, 0.f, 0.f, 0.f, 0.f, 0.f, 0.f, 0.f};
#pragma unroll
  for (int i = 0; i < 4; ++i) acc[i] = zero;

#if HAVE_TDM
  if (wave == 0) {
    tdm_load_2d((unsigned int)(size_t)&Bsh[0][0][0],
                (unsigned long long)(size_t)(W + (size_t)n0 * D_MODEL),
                D_MODEL, D_MODEL, KSTEP, 64, D_MODEL);
    __builtin_amdgcn_s_wait_tensorcnt(0);
  }
  __syncthreads();
#else
  for (int idx = tid; idx < 64 * (KSTEP / 8); idx += 256) {
    int row = idx >> 3, c = idx & 7;
    *(v8h*)&Bsh[0][row][c * 8] =
        *(const v8h*)&W[(size_t)(n0 + row) * D_MODEL + c * 8];
  }
  __syncthreads();
#endif

  for (int k0 = 0; k0 < D_MODEL; k0 += KSTEP) {
    const int buf = (k0 >> 6) & 1;
    if (k0 + KSTEP < D_MODEL) {
#if HAVE_TDM
      if (wave == 0) {
        tdm_load_2d((unsigned int)(size_t)&Bsh[buf ^ 1][0][0],
                    (unsigned long long)(size_t)(W + (size_t)n0 * D_MODEL +
                                                 (k0 + KSTEP)),
                    D_MODEL, D_MODEL, KSTEP, 64, D_MODEL);
      }
#else
      for (int idx = tid; idx < 64 * (KSTEP / 8); idx += 256) {
        int row = idx >> 3, c = idx & 7;
        *(v8h*)&Bsh[buf ^ 1][row][c * 8] =
            *(const v8h*)&W[(size_t)(n0 + row) * D_MODEL + (k0 + KSTEP) + c * 8];
      }
#endif
    }

    // ---- hoist all fragment loads for this 64-wide k-tile ----
    v16h a[2];
    v16h b[2][4];
#pragma unroll
    for (int ks = 0; ks < 2; ++ks) {
      const int kk = k0 + ks * 32;
      // A fragment: lanes 0-15 hold K {kk..kk+15}; lanes 16-31 +16
      a[ks] = *(const v16h*)(arow + kk + (hi ? 16 : 0));
#pragma unroll
      for (int nt = 0; nt < 4; ++nt) {
        // B fragment from LDS: lane holds column n0+nt*16+l16, contiguous K
        b[ks][nt] =
            *(const v16h*)&Bsh[buf][nt * 16 + l16][ks * 32 + (hi ? 16 : 0)];
      }
    }
    // ---- 8 back-to-back WMMAs ----
#pragma unroll
    for (int ks = 0; ks < 2; ++ks) {
#pragma unroll
      for (int nt = 0; nt < 4; ++nt) {
        acc[nt] = wmma16(a[ks], b[ks][nt], acc[nt]);
      }
    }

#if HAVE_TDM
    if (wave == 0) __builtin_amdgcn_s_wait_tensorcnt(0);
#endif
    __syncthreads();
  }

#pragma unroll
  for (int nt = 0; nt < 4; ++nt) {
#pragma unroll
    for (int r = 0; r < 8; ++r) {
      const int m = m0 + r + (hi ? 8 : 0);     // C layout: lanes16-31 -> M+8
      const int n = n0 + nt * 16 + l16;
      const float v = acc[nt][r];
      if (mode == 3) {
        outF[(size_t)m * D_MODEL + n] = v;
      } else if (mode == 0) {
        outH[(size_t)m * D_MODEL + n] = (h16)v;
      } else {
        const int b2 = m / SEQ, s = m % SEQ, h = n >> 6, d = n & 63;
        const size_t idx =
            (mode == 1)
                ? (((size_t)(b2 * N_HEADS + h) * SEQ + s) * D_K + d)
                : (((size_t)(b2 * N_HEADS + h) * D_K + d) * SEQ + s);
        outH[idx] = (h16)v;
      }
    }
  }
}

// ---------------------------------------------------------------------------
// Flash-style attention. grid = (SEQ/64, B*H), block = 128 (4 waves).
// Each wave owns 16 query rows, iterates key tiles of 32, online softmax.
// Q,K: [B,H,S,64]  Vt: [B,H,64,S]  A: [B,S,D_MODEL] (merged heads, f16)
// ---------------------------------------------------------------------------
__global__ __launch_bounds__(128) void attn_kernel(
    const h16* __restrict__ Q, const h16* __restrict__ K,
    const h16* __restrict__ Vt, h16* __restrict__ A) {
  __shared__ __align__(16) h16 Pbuf[4][16][32];  // per-wave P tile (f16)

  const int lane = threadIdx.x & 31;
  const int wave = threadIdx.x >> 5;
  const int l16  = lane & 15;
  const int hi   = (lane >> 4) & 1;

  const int bh = blockIdx.y;          // b*16 + h
  const int b  = bh >> 4;
  const int h  = bh & 15;
  const int q0 = blockIdx.x * 64 + wave * 16;

  // Q A-fragments for d_k chunks [0,32) and [32,64)
  const h16* Qrow = Q + ((size_t)bh * SEQ + q0 + l16) * D_K;
  const v16h Qa0 = *(const v16h*)(Qrow + (hi ? 16 : 0));
  const v16h Qa1 = *(const v16h*)(Qrow + 32 + (hi ? 16 : 0));

  float m_i[8], l_i[8];
#pragma unroll
  for (int r = 0; r < 8; ++r) { m_i[r] = -3.0e38f; l_i[r] = 0.f; }

  v8f O[4];
  const v8f zero = {0.f, 0.f, 0.f, 0.f, 0.f, 0.f, 0.f, 0.f};
#pragma unroll
  for (int i = 0; i < 4; ++i) O[i] = zero;

  for (int kt = 0; kt < SEQ / 32; ++kt) {
    // --- scores = Q * K^T for 32 keys (two 16-key column tiles) ---
    const h16* Kbase = K + ((size_t)bh * SEQ + (size_t)kt * 32) * D_K;
    const v16h Kb00 = *(const v16h*)(Kbase + (size_t)l16 * D_K + (hi ? 16 : 0));
    const v16h Kb01 = *(const v16h*)(Kbase + (size_t)l16 * D_K + 32 + (hi ? 16 : 0));
    const v16h Kb10 = *(const v16h*)(Kbase + (size_t)(16 + l16) * D_K + (hi ? 16 : 0));
    const v16h Kb11 = *(const v16h*)(Kbase + (size_t)(16 + l16) * D_K + 32 + (hi ? 16 : 0));

    v8f s0 = wmma16(Qa0, Kb00, zero);
    s0 = wmma16(Qa1, Kb01, s0);
    v8f s1 = wmma16(Qa0, Kb10, zero);
    s1 = wmma16(Qa1, Kb11, s1);

    // --- online softmax (row = accumulator element r + lane-half) ---
#pragma unroll
    for (int r = 0; r < 8; ++r) {
      float a0 = s0[r] * 0.125f;   // 1/sqrt(64)
      float a1 = s1[r] * 0.125f;
      float mx = fmaxf(a0, a1);
      mx = fmaxf(mx, __shfl_xor(mx, 1, 32));
      mx = fmaxf(mx, __shfl_xor(mx, 2, 32));
      mx = fmaxf(mx, __shfl_xor(mx, 4, 32));
      mx = fmaxf(mx, __shfl_xor(mx, 8, 32));
      const float mnew  = fmaxf(m_i[r], mx);
      const float alpha = __expf(m_i[r] - mnew);
      const float p0 = __expf(a0 - mnew);
      const float p1 = __expf(a1 - mnew);
      float ps = p0 + p1;
      ps += __shfl_xor(ps, 1, 32);
      ps += __shfl_xor(ps, 2, 32);
      ps += __shfl_xor(ps, 4, 32);
      ps += __shfl_xor(ps, 8, 32);
      l_i[r] = l_i[r] * alpha + ps;
      m_i[r] = mnew;
#pragma unroll
      for (int nt = 0; nt < 4; ++nt) O[nt][r] *= alpha;

      const int row = r + (hi ? 8 : 0);
      Pbuf[wave][row][l16]      = (h16)p0;
      Pbuf[wave][row][16 + l16] = (h16)p1;
    }
    __syncthreads();

    // Re-shape P (C layout) -> A fragment via LDS
  const v16h Pa = combine8(*(const v8h*)&Pbuf[wave][l16][(hi ? 16 : 0)],
                             *(const v8h*)&Pbuf[wave][l16][8 + (hi ? 16 : 0)]);

    // --- O += P * V : B fragment from transposed V, keys contiguous ---
    v16h Vb[4];
#pragma unroll
    for (int nt = 0; nt < 4; ++nt) {
      const h16* Vrow = Vt + ((size_t)bh * D_K + nt * 16 + l16) * SEQ +
                        (size_t)kt * 32 + (hi ? 16 : 0);
      Vb[nt] = *(const v16h*)Vrow;
    }
#pragma unroll
    for (int nt = 0; nt < 4; ++nt) O[nt] = wmma16(Pa, Vb[nt], O[nt]);
    __syncthreads();
  }

  // normalize and store merged-head output [B,S,D_MODEL]
#pragma unroll
  for (int r = 0; r < 8; ++r) {
    const float inv = 1.0f / l_i[r];
    const int q = q0 + r + (hi ? 8 : 0);
#pragma unroll
    for (int nt = 0; nt < 4; ++nt) {
      A[((size_t)b * SEQ + q) * D_MODEL + h * D_K + nt * 16 + l16] =
          (h16)(O[nt][r] * inv);
    }
  }
}

// ---------------------------------------------------------------------------
// launch
// ---------------------------------------------------------------------------
extern "C" void kernel_launch(void* const* d_in, const int* in_sizes, int n_in,
                              void* d_out, int out_size, void* d_ws,
                              size_t ws_size, hipStream_t stream) {
  const float* f_query = (const float*)d_in[0];
  const float* f_key   = (const float*)d_in[1];
  const float* f_value = (const float*)d_in[2];
  const float* f_Wq    = (const float*)d_in[3];
  const float* f_Wk    = (const float*)d_in[4];
  const float* f_Wv    = (const float*)d_in[5];
  const float* f_Wo    = (const float*)d_in[6];

  uint8_t* ws = (uint8_t*)d_ws;
  const size_t WB = (size_t)D_MODEL * D_MODEL * sizeof(h16);  // 2 MB
  const size_t XB = (size_t)M_ROWS * D_MODEL * sizeof(h16);   // 16 MB

  h16* Wq16 = (h16*)(ws + 0 * WB);
  h16* Wk16 = (h16*)(ws + 1 * WB);
  h16* Wv16 = (h16*)(ws + 2 * WB);
  h16* Wo16 = (h16*)(ws + 3 * WB);
  uint8_t* p = ws + 4 * WB;
  h16* Xq16 = (h16*)p; p += XB;
  h16* Xk16 = (h16*)p; p += XB;
  h16* Xv16 = (h16*)p; p += XB;
  h16* Q16  = (h16*)p; p += XB;   // [B,H,S,64]
  h16* K16  = (h16*)p; p += XB;   // [B,H,S,64]
  h16* Vt16 = (h16*)p; p += XB;   // [B,H,64,S]
  h16* A16  = (h16*)p; p += XB;   // [B,S,D_MODEL]

  const int NX = M_ROWS * D_MODEL;    // 8388608
  const int NW = D_MODEL * D_MODEL;   // 1048576
  cvt_f32_to_f16<<<2048, 256, 0, stream>>>(f_query, Xq16, NX);
  cvt_f32_to_f16<<<2048, 256, 0, stream>>>(f_key,   Xk16, NX);
  cvt_f32_to_f16<<<2048, 256, 0, stream>>>(f_value, Xv16, NX);
  cvt_f32_to_f16<<<1024, 256, 0, stream>>>(f_Wq, Wq16, NW);
  cvt_f32_to_f16<<<1024, 256, 0, stream>>>(f_Wk, Wk16, NW);
  cvt_f32_to_f16<<<1024, 256, 0, stream>>>(f_Wv, Wv16, NW);
  cvt_f32_to_f16<<<1024, 256, 0, stream>>>(f_Wo, Wo16, NW);

  const dim3 ggrid(D_MODEL / 64, M_ROWS / 128);  // (16, 64)
  gemm_xwT<<<ggrid, 256, 0, stream>>>(Xq16, Wq16, Q16,  nullptr, 1);
  gemm_xwT<<<ggrid, 256, 0, stream>>>(Xk16, Wk16, K16,  nullptr, 1);
  gemm_xwT<<<ggrid, 256, 0, stream>>>(Xv16, Wv16, Vt16, nullptr, 2);

  attn_kernel<<<dim3(SEQ / 64, B_SZ * N_HEADS), 128, 0, stream>>>(Q16, K16,
                                                                  Vt16, A16);

  gemm_xwT<<<ggrid, 256, 0, stream>>>(A16, Wo16, nullptr, (float*)d_out, 3);
}
